// ProteinAtomicChimeraEmbedder_63608465654111
// MI455X (gfx1250) — matrix operand order repaired and checked by reading, not compiled, not run on hardware
//
#include <hip/hip_runtime.h>
#include <hip/hip_bf16.h>
#include <math.h>

// ---------------------------------------------------------------------------
// Problem constants (from the reference)
// ---------------------------------------------------------------------------
#define N_ATOM   14336
#define N_RES    1024
#define E_ATOM_N 131072
#define E_RES_N  30720
#define C_S      128
#define C_Z      128
#define X_DIM    56      // atom irrep feature width  (16 + 12 + 12 + 16)
#define RU_DIM   112     // residue irrep width       (32 + 24 + 24 + 32)

typedef __attribute__((ext_vector_type(16))) _Float16 v16h;
typedef __attribute__((ext_vector_type(8)))  float    v8f;

// ---------------------------------------------------------------------------
// WMMA helpers (wave32, CDNA5 layouts per CDNA5 ISA)
// ---------------------------------------------------------------------------
__device__ __forceinline__ v8f wmma_f16(v16h a, v16h b, v8f c) {
  // v_wmma_f32_16x16x32_f16 : D = A(16x32 f16) * B(32x16 f16) + C(16x16 f32)
  return __builtin_amdgcn_wmma_f32_16x16x32_f16(
      /*neg_a=*/false, a, /*neg_b=*/false, b,
      /*c_mod=*/(short)0, c, /*reuse_a=*/false, /*reuse_b=*/false);
}

// A-matrix (16x32 f16) from row-major f16 LDS tile (no bounds checks).
// lane L: M = L&15 ; element j holds K = (j&7) + 8*(L>=16) + 16*(j>=8)
__device__ __forceinline__ v16h lds_load_A(const _Float16* base, int ld, int k0) {
  int lane = threadIdx.x & 31;
  int m  = lane & 15;
  int kb = (lane >> 4) << 3;
  v16h a;
#pragma unroll
  for (int j = 0; j < 16; ++j) {
    int k = k0 + (j & 7) + kb + ((j >> 3) << 4);
    a[j] = base[m * ld + k];
  }
  return a;
}

// A-matrix from row-major f32 global memory; K must be in-bounds (no checks).
// Two-phase (load-all, then convert) so the compiler batches the loads.
__device__ __forceinline__ v16h gmem_load_A(const float* A, int lda, int m0, int k0) {
  int lane = threadIdx.x & 31;
  int m  = m0 + (lane & 15);
  int kb = (lane >> 4) << 3;
  float tmp[16];
#pragma unroll
  for (int j = 0; j < 16; ++j) {
    int k = k0 + (j & 7) + kb + ((j >> 3) << 4);
    tmp[j] = A[(size_t)m * lda + k];
  }
  v16h a;
#pragma unroll
  for (int j = 0; j < 16; ++j) a[j] = (_Float16)tmp[j];
  return a;
}

// B-matrix (32x16 f16) from row-major f32 global (K x N).
// Branch-free K bound handling: clamped address (always legal) + select-zero.
// lane L: N = L&15 ; element j holds K = (j&7) + 8*(L>=16) + 16*(j>=8)
__device__ __forceinline__ v16h gmem_load_B(const float* B, int ldb, int k0,
                                            int n0, int kmax) {
  int lane = threadIdx.x & 31;
  int n  = n0 + (lane & 15);
  int kb = k0 + ((lane >> 4) << 3);
  float tmp[16];
#pragma unroll
  for (int j = 0; j < 16; ++j) {
    int k  = kb + (j & 7) + ((j >> 3) << 4);
    int kc = (k < kmax) ? k : (kmax - 1);        // in-bounds address
    tmp[j] = B[(size_t)kc * ldb + n];
  }
  v16h b;
#pragma unroll
  for (int j = 0; j < 16; ++j) {
    int k = kb + (j & 7) + ((j >> 3) << 4);
    b[j] = (k < kmax) ? (_Float16)tmp[j] : (_Float16)0.f;  // v_cndmask, no branch
  }
  return b;
}

// C/D layout: lane L -> N = L&15 ; vgpr r -> M = r + 8*(L>=16)
#define C_ROW(lane, r) (((lane >> 4) << 3) + (r))
#define C_COL(lane)    ((lane) & 15)

__device__ __forceinline__ void quat_rot(const float* q4, float R[3][3]) {
  float w = q4[0], x = q4[1], y = q4[2], z = q4[3];
  float n = rsqrtf(w * w + x * x + y * y + z * z + 1e-12f);
  w *= n; x *= n; y *= n; z *= n;
  R[0][0] = 1.f - 2.f * (y * y + z * z); R[0][1] = 2.f * (x * y - w * z); R[0][2] = 2.f * (x * z + w * y);
  R[1][0] = 2.f * (x * y + w * z); R[1][1] = 1.f - 2.f * (x * x + z * z); R[1][2] = 2.f * (y * z - w * x);
  R[2][0] = 2.f * (x * z - w * y); R[2][1] = 2.f * (y * z + w * x); R[2][2] = 1.f - 2.f * (x * x + y * y);
}

// ---------------------------------------------------------------------------
// Fused edge kernel:  w = mlp2(attr)  (WMMA, weight tile kept in LDS)
//                     out_accum[dst] += tensor_product(x[src], sh, w)
// Input irreps fixed at ATOM_MULS=(16,4,4,16); output muls are template args.
// One workgroup (128 thr = 4 waves) handles 16 edges.
// ---------------------------------------------------------------------------
template <int B0, int B1, int B2, int B3, bool GATHER>
__global__ __launch_bounds__(128)
void k_edge_tp(const float* __restrict__ attr,   // (E,16)
               const float* __restrict__ sh,     // (E,4)
               const float* __restrict__ xfeat,  // (*,56)
               const int*   __restrict__ src_idx,// (E) or null
               const int*   __restrict__ dst_idx,// (E)
               const float* __restrict__ w1, const float* __restrict__ b1,
               const float* __restrict__ w2, const float* __restrict__ b2,
               float* __restrict__ accum, float* __restrict__ cnt) {
  // weight-block offsets inside flat weight vector (path order of reference)
  constexpr int P0 = 0;
  constexpr int P1 = P0 + 16 * B0;
  constexpr int P2 = P1 + 16 * B1;
  constexpr int P3 = P2 + 4 * B1;
  constexpr int P4 = P3 + 4 * B0;
  constexpr int P5 = P4 + 4 * B2;
  constexpr int P6 = P5 + 4 * B2;
  constexpr int P7 = P6 + 4 * B3;
  constexpr int P8 = P7 + 4 * B1;
  constexpr int P9 = P8 + 16 * B3;
  constexpr int NE = P9 + 16 * B2;        // 832 (atom) / 1664 (agg)
  constexpr int NT = NE / 16;             // N-tiles for the weight GEMM
  constexpr int O1 = B0;
  constexpr int O2 = B0 + 3 * B1;
  constexpr int O3 = B0 + 3 * (B1 + B2);
  constexpr int OUT = O3 + B3;            // 56 / 112
  constexpr int UNITS = B0 + B1 + B2 + B3;

  __shared__ _Float16 sAttr[16][32];      // K padded 16 -> 32
  __shared__ _Float16 sHid[16][32];
  __shared__ _Float16 sW[16][NE];         // per-edge weight tile (LDS resident!)
  __shared__ float    sX[16][X_DIM];
  __shared__ float    sSh[16][4];
  __shared__ int      sSrc[16], sDst[16];

  const int e0   = blockIdx.x * 16;
  const int t    = threadIdx.x;
  const int wave = t >> 5;
  const int lane = t & 31;

  if (t < 16) {
    sDst[t] = dst_idx[e0 + t];
    sSrc[t] = GATHER ? src_idx[e0 + t] : (e0 + t);
  }
  __syncthreads();

  __builtin_prefetch(w2, 0, 1);  // weight matrix streamed by every WG

  for (int i = t; i < 16 * 32; i += 128) {
    int r = i >> 5, k = i & 31;
    sAttr[r][k] = (k < 16) ? (_Float16)attr[(size_t)(e0 + r) * 16 + k]
                           : (_Float16)0.f;
  }
  for (int i = t; i < 16 * X_DIM; i += 128) {
    int r = i / X_DIM, c = i % X_DIM;
    sX[r][c] = xfeat[(size_t)sSrc[r] * X_DIM + c];
  }
  for (int i = t; i < 16 * 4; i += 128) {
    int r = i >> 2, c = i & 3;
    sSh[r][c] = sh[(size_t)(e0 + r) * 4 + c];
  }
  __syncthreads();

  // hidden = relu(attr @ w1 + b1) : one WMMA, wave 0
  if (wave == 0) {
    v16h a = lds_load_A(&sAttr[0][0], 32, 0);
    v16h b = gmem_load_B(w1, 16, 0, 0, 16);
    v8f  c = {};
    c = wmma_f16(a, b, c);
    int n = C_COL(lane);
#pragma unroll
    for (int r = 0; r < 8; ++r) {
      int m = C_ROW(lane, r);
      sHid[m][n]      = (_Float16)fmaxf(c[r] + b1[n], 0.f);
      sHid[m][n + 16] = (_Float16)0.f;  // K padding
    }
  }
  __syncthreads();

  // w = hidden @ w2 + b2  -> LDS, f16.  4 waves x (NT/4) tiles
  for (int nt = wave; nt < NT; nt += 4) {
    int n0 = nt * 16;
    v16h a = lds_load_A(&sHid[0][0], 32, 0);
    v16h b = gmem_load_B(w2, NE, 0, n0, 16);
    v8f  c = {};
    c = wmma_f16(a, b, c);
    int n = n0 + C_COL(lane);
#pragma unroll
    for (int r = 0; r < 8; ++r)
      sW[C_ROW(lane, r)][n] = (_Float16)(c[r] + b2[n]);
  }
  __syncthreads();

  // ---- tensor product, scatter-add -------------------------------------
  for (int u = t; u < 16 * UNITS; u += 128) {
    int e = u / UNITS, q = u % UNITS;
    const _Float16* w = sW[e];
    const float*    x = sX[e];
    float h0 = sSh[e][0], hx = sSh[e][1], hy = sSh[e][2], hz = sSh[e][3];
    float* out = accum + (size_t)sDst[e] * OUT;

    if (q < B0) {                               // out0e
      float a = 0.f;
      for (int i = 0; i < 16; ++i) a += (float)w[P0 + i * B0 + q] * x[i];
      a *= h0;
      for (int i = 0; i < 4; ++i) {
        float dv = x[16 + 3 * i] * hx + x[17 + 3 * i] * hy + x[18 + 3 * i] * hz;
        a += (float)w[P3 + i * B0 + q] * dv;
      }
      atomicAdd(out + q, a);
    } else if (q < B0 + B1) {                   // out1o (vector)
      int o = q - B0;
      float t1 = 0.f, ax = 0.f, ay = 0.f, az = 0.f, cx = 0.f, cy = 0.f, cz = 0.f;
      for (int i = 0; i < 16; ++i) t1 += (float)w[P1 + i * B1 + o] * x[i];
      for (int i = 0; i < 4; ++i) {
        float w2v = (float)w[P2 + i * B1 + o];
        ax += w2v * x[16 + 3 * i]; ay += w2v * x[17 + 3 * i]; az += w2v * x[18 + 3 * i];
        float vx = x[28 + 3 * i], vy = x[29 + 3 * i], vz = x[30 + 3 * i]; // v1e
        float w7v = (float)w[P7 + i * B1 + o];
        cx += w7v * (vy * hz - vz * hy);
        cy += w7v * (vz * hx - vx * hz);
        cz += w7v * (vx * hy - vy * hx);
      }
      atomicAdd(out + O1 + 3 * o + 0, t1 * hx + ax * h0 + cx);
      atomicAdd(out + O1 + 3 * o + 1, t1 * hy + ay * h0 + cy);
      atomicAdd(out + O1 + 3 * o + 2, t1 * hz + az * h0 + cz);
    } else if (q < B0 + B1 + B2) {              // out1e (vector)
      int o = q - B0 - B1;
      float cx = 0.f, cy = 0.f, cz = 0.f, ax = 0.f, ay = 0.f, az = 0.f, t9 = 0.f;
      for (int i = 0; i < 4; ++i) {
        float vx = x[16 + 3 * i], vy = x[17 + 3 * i], vz = x[18 + 3 * i]; // v1o
        float w4v = (float)w[P4 + i * B2 + o];
        cx += w4v * (vy * hz - vz * hy);
        cy += w4v * (vz * hx - vx * hz);
        cz += w4v * (vx * hy - vy * hx);
        float w5v = (float)w[P5 + i * B2 + o];
        ax += w5v * x[28 + 3 * i]; ay += w5v * x[29 + 3 * i]; az += w5v * x[30 + 3 * i];
      }
      for (int i = 0; i < 16; ++i) t9 += (float)w[P9 + i * B2 + o] * x[40 + i];
      atomicAdd(out + O2 + 3 * o + 0, cx + ax * h0 + t9 * hx);
      atomicAdd(out + O2 + 3 * o + 1, cy + ay * h0 + t9 * hy);
      atomicAdd(out + O2 + 3 * o + 2, cz + az * h0 + t9 * hz);
    } else {                                    // out0o
      int o = q - B0 - B1 - B2;
      float a = 0.f, a2 = 0.f;
      for (int i = 0; i < 4; ++i) {
        float dv = x[28 + 3 * i] * hx + x[29 + 3 * i] * hy + x[30 + 3 * i] * hz;
        a += (float)w[P6 + i * B3 + o] * dv;
      }
      for (int i = 0; i < 16; ++i) a2 += (float)w[P8 + i * B3 + o] * x[40 + i];
      atomicAdd(out + O3 + o, a + a2 * h0);
    }
  }
  if (t < 16) atomicAdd(cnt + sDst[t], 1.0f);
}

// ---------------------------------------------------------------------------
// atom_out = atom_features + accum / max(cnt,1)
// ---------------------------------------------------------------------------
__global__ void k_atom_out(const float* __restrict__ feat,
                           const float* __restrict__ accum,
                           const float* __restrict__ cnt,
                           float* __restrict__ out, int total) {
  int i = blockIdx.x * blockDim.x + threadIdx.x;
  if (i >= total) return;
  int a = i / X_DIM;
  out[i] = feat[i] + accum[i] / fmaxf(cnt[a], 1.0f);
}

// ---------------------------------------------------------------------------
// res_update = wigner_rotate(res_accum / max(cnt,1), R^T)   (RES_MULS layout)
// Output buffer is K-padded to 128 columns (cols 112..127 = 0) so the
// following GEMM chain runs with a compile-time K that is a multiple of 32.
// ---------------------------------------------------------------------------
__global__ __launch_bounds__(128)
void k_res_update(const float* __restrict__ accum, const float* __restrict__ cnt,
                  const float* __restrict__ quats, float* __restrict__ ru) {
  int r = blockIdx.x, t = threadIdx.x;
  float val = 0.f;
  if (t < RU_DIM) {
    float c = fmaxf(cnt[r], 1.0f);
    const float* a = accum + (size_t)r * RU_DIM;
    if (t >= 32 && t < 80) {           // vector irreps: rotate by Rinv = R^T
      int comp = (t - 32) % 3;
      int vb = t - comp;
      float R[3][3];
      quat_rot(quats + (size_t)r * 4, R);
      float v0 = a[vb] / c, v1 = a[vb + 1] / c, v2 = a[vb + 2] / c;
      val = R[0][comp] * v0 + R[1][comp] * v1 + R[2][comp] * v2;
    } else {
      val = a[t] / c;
    }
  }
  ru[(size_t)r * C_S + t] = val;       // zero padding for t in [112,128)
}

// ---------------------------------------------------------------------------
// Generic WMMA GEMM: C = [relu](A(MxK) @ B(KB x N) + bias)
// K: compile-time, multiple of 32 (A fully in-bounds).
// KB: real rows of B (KB <= K); KB==K folds the clamp away at compile time.
// One wave per 16x16 output tile.
// ---------------------------------------------------------------------------
template <int K, int KB, int RELU>
__global__ __launch_bounds__(32)
void k_gemm(const float* __restrict__ A, const float* __restrict__ B,
            const float* __restrict__ bias, float* __restrict__ C, int N) {
  int tiles_n = N >> 4;
  int mt = blockIdx.x / tiles_n, nt = blockIdx.x % tiles_n;
  int m0 = mt << 4, n0 = nt << 4;
  v8f acc = {};
#pragma unroll
  for (int k0 = 0; k0 < K; k0 += 32) {
    v16h a = gmem_load_A(A, K, m0, k0);
    v16h b = gmem_load_B(B, N, k0, n0, KB);
    acc = wmma_f16(a, b, acc);
  }
  int lane = threadIdx.x & 31;
  int n = n0 + C_COL(lane);
#pragma unroll
  for (int r = 0; r < 8; ++r) {
    float v = acc[r] + bias[n];
    if (RELU) v = fmaxf(v, 0.f);
    C[(size_t)(m0 + C_ROW(lane, r)) * N + n] = v;
  }
}

// ---------------------------------------------------------------------------
// s1 = layernorm(res_features + h3) ; pts = s1@pt_w+pt_b ; pts_g = R pts + t
// ---------------------------------------------------------------------------
__global__ __launch_bounds__(128)
void k_s1_pts(const float* __restrict__ resf, const float* __restrict__ h3,
              const float* __restrict__ g, const float* __restrict__ bb,
              const float* __restrict__ ptw, const float* __restrict__ ptb,
              const float* __restrict__ quats, const float* __restrict__ trans,
              float* __restrict__ s1, float* __restrict__ ptsg) {
  __shared__ float red[8];
  __shared__ float row[C_S];
  __shared__ float pts[24];
  int r = blockIdx.x, t = threadIdx.x;
  float v = resf[(size_t)r * C_S + t] + h3[(size_t)r * C_S + t];
  float sum = v;
  for (int o = 16; o; o >>= 1) sum += __shfl_xor(sum, o, 32);
  if ((t & 31) == 0) red[t >> 5] = sum;
  __syncthreads();
  float mu = (red[0] + red[1] + red[2] + red[3]) * (1.f / 128.f);
  float d = v - mu, sq = d * d;
  for (int o = 16; o; o >>= 1) sq += __shfl_xor(sq, o, 32);
  if ((t & 31) == 0) red[4 + (t >> 5)] = sq;
  __syncthreads();
  float var = (red[4] + red[5] + red[6] + red[7]) * (1.f / 128.f);
  float sv = d * rsqrtf(var + 1e-5f) * g[t] + bb[t];
  s1[(size_t)r * C_S + t] = sv;
  row[t] = sv;
  __syncthreads();
  if (t < 24) {
    float acc = ptb[t];
    for (int k = 0; k < C_S; ++k) acc += row[k] * ptw[k * 24 + t];
    pts[t] = acc;
  }
  __syncthreads();
  if (t < 8) {
    float R[3][3];
    quat_rot(quats + (size_t)r * 4, R);
    float px = pts[3 * t], py = pts[3 * t + 1], pz = pts[3 * t + 2];
    ptsg[(size_t)r * 24 + 3 * t + 0] = R[0][0] * px + R[0][1] * py + R[0][2] * pz + trans[r * 3 + 0];
    ptsg[(size_t)r * 24 + 3 * t + 1] = R[1][0] * px + R[1][1] * py + R[1][2] * pz + trans[r * 3 + 1];
    ptsg[(size_t)r * 24 + 3 * t + 2] = R[2][0] * px + R[2][1] * py + R[2][2] * pz + trans[r * 3 + 2];
  }
}

// ---------------------------------------------------------------------------
// Fused residue-edge message block: 16 edges per workgroup.
//   m_in = [s1[rd], s1[rs], z, dist]  (LDS, f16, K padded 392->416)
//   m    = relu(m_in@m_w1+b1)@m_w2+b2          (WMMA)
//   msum[rd] += m ; mcnt[rd] += 1
//   z_out = layernorm(z + m@zu_w + zu_b)       (WMMA + in-LDS layernorm)
// ---------------------------------------------------------------------------
__global__ __launch_bounds__(128)
void k_res_edge(const int* __restrict__ reidx, const float* __restrict__ s1,
                const float* __restrict__ ptsg, const float* __restrict__ zf,
                const float* __restrict__ m_w1, const float* __restrict__ m_b1,
                const float* __restrict__ m_w2, const float* __restrict__ m_b2,
                const float* __restrict__ zu_w, const float* __restrict__ zu_b,
                const float* __restrict__ lnz_g, const float* __restrict__ lnz_b,
                float* __restrict__ msum, float* __restrict__ mcnt,
                float* __restrict__ zout) {
  __shared__ _Float16 sIn[16][416];
  __shared__ _Float16 sH[16][C_S];
  __shared__ float    sM[16][C_S];
  __shared__ _Float16 sMh[16][C_S];
  __shared__ float    sZ[16][C_Z];
  __shared__ int      sRd[16], sRs[16];

  const int e0 = blockIdx.x * 16;
  const int t = threadIdx.x, wave = t >> 5, lane = t & 31;

  if (t < 16) { sRd[t] = reidx[e0 + t]; sRs[t] = reidx[E_RES_N + e0 + t]; }
  __syncthreads();

  __builtin_prefetch(m_w1, 0, 1);

  for (int i = t; i < 16 * 416; i += 128) {
    int r = i / 416, c = i % 416;
    int rd = sRd[r], rs = sRs[r];
    float v;
    if (c < 128)       v = s1[(size_t)rd * C_S + c];
    else if (c < 256)  v = s1[(size_t)rs * C_S + (c - 128)];
    else if (c < 384)  v = zf[(size_t)(e0 + r) * C_Z + (c - 256)];
    else if (c < 392) {
      int n = c - 384;
      float dx = ptsg[rd * 24 + 3 * n + 0] - ptsg[rs * 24 + 3 * n + 0];
      float dy = ptsg[rd * 24 + 3 * n + 1] - ptsg[rs * 24 + 3 * n + 1];
      float dz = ptsg[rd * 24 + 3 * n + 2] - ptsg[rs * 24 + 3 * n + 2];
      v = sqrtf(dx * dx + dy * dy + dz * dz + 1e-8f);
    } else v = 0.f;  // K padding
    sIn[r][c] = (_Float16)v;
  }
  __syncthreads();

  // h = relu(m_in @ m_w1 + b1), K = 392 (padded 416)
  for (int nt = wave; nt < 8; nt += 4) {
    v8f acc = {};
#pragma unroll
    for (int k0 = 0; k0 < 416; k0 += 32) {
      v16h a = lds_load_A(&sIn[0][0], 416, k0);
      v16h b = gmem_load_B(m_w1, C_S, k0, nt * 16, 392);
      acc = wmma_f16(a, b, acc);
    }
    int n = nt * 16 + C_COL(lane);
#pragma unroll
    for (int r = 0; r < 8; ++r)
      sH[C_ROW(lane, r)][n] = (_Float16)fmaxf(acc[r] + m_b1[n], 0.f);
  }
  __syncthreads();

  // m = h @ m_w2 + b2
  for (int nt = wave; nt < 8; nt += 4) {
    v8f acc = {};
#pragma unroll
    for (int k0 = 0; k0 < 128; k0 += 32) {
      v16h a = lds_load_A(&sH[0][0], C_S, k0);
      v16h b = gmem_load_B(m_w2, C_S, k0, nt * 16, 128);
      acc = wmma_f16(a, b, acc);
    }
    int n = nt * 16 + C_COL(lane);
#pragma unroll
    for (int r = 0; r < 8; ++r) {
      float v = acc[r] + m_b2[n];
      sM[C_ROW(lane, r)][n]  = v;
      sMh[C_ROW(lane, r)][n] = (_Float16)v;
    }
  }
  __syncthreads();

  // scatter segment-sum of m by rd
  for (int i = t; i < 16 * C_S; i += 128) {
    int r = i >> 7, c = i & 127;
    atomicAdd(&msum[(size_t)sRd[r] * C_S + c], sM[r][c]);
  }
  if (t < 16) atomicAdd(&mcnt[sRd[t]], 1.0f);

  // z = zf + m @ zu_w + zu_b
  for (int nt = wave; nt < 8; nt += 4) {
    v8f acc = {};
#pragma unroll
    for (int k0 = 0; k0 < 128; k0 += 32) {
      v16h a = lds_load_A(&sMh[0][0], C_S, k0);
      v16h b = gmem_load_B(zu_w, C_Z, k0, nt * 16, 128);
      acc = wmma_f16(a, b, acc);
    }
    int n = nt * 16 + C_COL(lane);
#pragma unroll
    for (int r = 0; r < 8; ++r) {
      int row = C_ROW(lane, r);
      sZ[row][n] = zf[(size_t)(e0 + row) * C_Z + n] + acc[r] + zu_b[n];
    }
  }
  __syncthreads();

  // per-row layernorm -> global z output
  if (t < 16) {
    float mu = 0.f;
    for (int c = 0; c < C_Z; ++c) mu += sZ[t][c];
    mu *= (1.f / 128.f);
    float var = 0.f;
    for (int c = 0; c < C_Z; ++c) { float d = sZ[t][c] - mu; var += d * d; }
    var *= (1.f / 128.f);
    float is = rsqrtf(var + 1e-5f);
    for (int c = 0; c < C_Z; ++c)
      zout[(size_t)(e0 + t) * C_Z + c] = (sZ[t][c] - mu) * is * lnz_g[c] + lnz_b[c];
  }
}

__global__ __launch_bounds__(128)
void k_mean(const float* __restrict__ msum, const float* __restrict__ mcnt,
            float* __restrict__ mmean) {
  int r = blockIdx.x, t = threadIdx.x;
  mmean[(size_t)r * C_S + t] = msum[(size_t)r * C_S + t] / fmaxf(mcnt[r], 1.0f);
}

__global__ __launch_bounds__(128)
void k_s_final(const float* __restrict__ s1, const float* __restrict__ sup,
               const float* __restrict__ g, const float* __restrict__ b,
               const float* __restrict__ mask, float* __restrict__ out) {
  __shared__ float red[8];
  int r = blockIdx.x, t = threadIdx.x;
  float v = s1[(size_t)r * C_S + t] + sup[(size_t)r * C_S + t];
  float sum = v;
  for (int o = 16; o; o >>= 1) sum += __shfl_xor(sum, o, 32);
  if ((t & 31) == 0) red[t >> 5] = sum;
  __syncthreads();
  float mu = (red[0] + red[1] + red[2] + red[3]) * (1.f / 128.f);
  float d = v - mu, sq = d * d;
  for (int o = 16; o; o >>= 1) sq += __shfl_xor(sq, o, 32);
  if ((t & 31) == 0) red[4 + (t >> 5)] = sq;
  __syncthreads();
  float var = (red[4] + red[5] + red[6] + red[7]) * (1.f / 128.f);
  out[(size_t)r * C_S + t] = (d * rsqrtf(var + 1e-5f) * g[t] + b[t]) * mask[r];
}

// ---------------------------------------------------------------------------
// Host launcher
// ---------------------------------------------------------------------------
extern "C" void kernel_launch(void* const* d_in, const int* in_sizes, int n_in,
                              void* d_out, int out_size, void* d_ws, size_t ws_size,
                              hipStream_t stream) {
  (void)in_sizes; (void)n_in; (void)out_size; (void)ws_size;

  const float* atom_features   = (const float*)d_in[0];
  const float* res_features    = (const float*)d_in[1];
  const float* res_edge_feat   = (const float*)d_in[2];
  const float* atom_edge_attr  = (const float*)d_in[3];
  const float* atom_edge_sh    = (const float*)d_in[4];
  const float* agg_edge_feat   = (const float*)d_in[5];
  const float* agg_edge_sh     = (const float*)d_in[6];
  const float* rigid_quats     = (const float*)d_in[7];
  const float* rigid_trans     = (const float*)d_in[8];
  const float* res_mask        = (const float*)d_in[9];
  const int*   atom_edge_index = (const int*)d_in[10];
  const int*   atom_res_batch  = (const int*)d_in[11];
  const int*   res_edge_index  = (const int*)d_in[12];
  // params, in dict order
  const float* ae_w1 = (const float*)d_in[13];
  const float* ae_b1 = (const float*)d_in[14];
  const float* ae_w2 = (const float*)d_in[15];
  const float* ae_b2 = (const float*)d_in[16];
  const float* ag_w1 = (const float*)d_in[17];
  const float* ag_b1 = (const float*)d_in[18];
  const float* ag_w2 = (const float*)d_in[19];
  const float* ag_b2 = (const float*)d_in[20];
  const float* i2s_w1 = (const float*)d_in[21];
  const float* i2s_b1 = (const float*)d_in[22];
  const float* i2s_w2 = (const float*)d_in[23];
  const float* i2s_b2 = (const float*)d_in[24];
  const float* i2s_w3 = (const float*)d_in[25];
  const float* i2s_b3 = (const float*)d_in[26];
  const float* ln1_g  = (const float*)d_in[27];
  const float* ln1_b  = (const float*)d_in[28];
  const float* pt_w   = (const float*)d_in[29];
  const float* pt_b   = (const float*)d_in[30];
  const float* m_w1   = (const float*)d_in[31];
  const float* m_b1   = (const float*)d_in[32];
  const float* m_w2   = (const float*)d_in[33];
  const float* m_b2   = (const float*)d_in[34];
  const float* su_w   = (const float*)d_in[35];
  const float* su_b   = (const float*)d_in[36];
  const float* lns_g  = (const float*)d_in[37];
  const float* lns_b  = (const float*)d_in[38];
  const float* zu_w   = (const float*)d_in[39];
  const float* zu_b   = (const float*)d_in[40];
  const float* lnz_g  = (const float*)d_in[41];
  const float* lnz_b  = (const float*)d_in[42];

  // output regions (f32, concatenated in tuple return order)
  float* out_atom = (float*)d_out;                       // N_ATOM*56
  float* out_s    = out_atom + (size_t)N_ATOM * X_DIM;   // N_RES*128
  float* out_z    = out_s + (size_t)N_RES * C_S;         // E_RES*128

  // workspace layout (zero-initialized region first)
  float* wsf = (float*)d_ws;
  size_t o = 0;
  float* atom_accum = wsf + o; o += (size_t)N_ATOM * X_DIM;
  float* atom_cnt   = wsf + o; o += N_ATOM;
  float* res_accum  = wsf + o; o += (size_t)N_RES * RU_DIM;
  float* res_cnt    = wsf + o; o += N_RES;
  float* msum       = wsf + o; o += (size_t)N_RES * C_S;
  float* mcnt       = wsf + o; o += N_RES;
  size_t zero_elems = o;
  float* res_upd = wsf + o; o += (size_t)N_RES * C_S;   // K-padded to 128
  float* h1      = wsf + o; o += (size_t)N_RES * C_S;
  float* h2      = wsf + o; o += (size_t)N_RES * C_S;
  float* h3      = wsf + o; o += (size_t)N_RES * C_S;
  float* s1buf   = wsf + o; o += (size_t)N_RES * C_S;
  float* ptsg    = wsf + o; o += (size_t)N_RES * 24;
  float* mmean   = wsf + o; o += (size_t)N_RES * C_S;
  float* sup     = wsf + o; o += (size_t)N_RES * C_S;

  hipMemsetAsync(d_ws, 0, zero_elems * sizeof(float), stream);

  // 1) atom-edge fused MLP + tensor product + scatter
  k_edge_tp<16, 4, 4, 16, true><<<E_ATOM_N / 16, 128, 0, stream>>>(
      atom_edge_attr, atom_edge_sh, atom_features,
      atom_edge_index + E_ATOM_N /*src*/, atom_edge_index /*dst*/,
      ae_w1, ae_b1, ae_w2, ae_b2, atom_accum, atom_cnt);

  // 2) atom_out = atom_features + mean
  k_atom_out<<<(N_ATOM * X_DIM + 255) / 256, 256, 0, stream>>>(
      atom_features, atom_accum, atom_cnt, out_atom, N_ATOM * X_DIM);

  // 3) atom->residue aggregation (fused MLP + TP + scatter)
  k_edge_tp<32, 8, 8, 32, false><<<N_ATOM / 16, 128, 0, stream>>>(
      agg_edge_feat, agg_edge_sh, out_atom,
      nullptr, atom_res_batch,
      ag_w1, ag_b1, ag_w2, ag_b2, res_accum, res_cnt);

  // 4) mean + Wigner rotate (writes K-padded 1024x128 buffer)
  k_res_update<<<N_RES, 128, 0, stream>>>(res_accum, res_cnt, rigid_quats, res_upd);

  // 5) i2s MLP chain (WMMA GEMMs, K=128 compile-time; first B has 112 rows)
  const int GEMM_GRID = (N_RES / 16) * (C_S / 16);
  k_gemm<128, 112, 1><<<GEMM_GRID, 32, 0, stream>>>(res_upd, i2s_w1, i2s_b1, h1, C_S);
  k_gemm<128, 128, 1><<<GEMM_GRID, 32, 0, stream>>>(h1, i2s_w2, i2s_b2, h2, C_S);
  k_gemm<128, 128, 0><<<GEMM_GRID, 32, 0, stream>>>(h2, i2s_w3, i2s_b3, h3, C_S);

  // 6) layernorm + points + rigid transform
  k_s1_pts<<<N_RES, 128, 0, stream>>>(res_features, h3, ln1_g, ln1_b, pt_w, pt_b,
                                      rigid_quats, rigid_trans, s1buf, ptsg);

  // 7) fused residue-edge message block (+ z output)
  k_res_edge<<<E_RES_N / 16, 128, 0, stream>>>(
      res_edge_index, s1buf, ptsg, res_edge_feat,
      m_w1, m_b1, m_w2, m_b2, zu_w, zu_b, lnz_g, lnz_b,
      msum, mcnt, out_z);

  // 8) s update: mean, GEMM, layernorm * mask
  k_mean<<<N_RES, 128, 0, stream>>>(msum, mcnt, mmean);
  k_gemm<128, 128, 0><<<GEMM_GRID, 32, 0, stream>>>(mmean, su_w, su_b, sup, C_S);
  k_s_final<<<N_RES, 128, 0, stream>>>(s1buf, sup, lns_g, lns_b, res_mask, out_s);
}
